// Block2_87144886436578
// MI455X (gfx1250) — matrix operand order repaired and checked by reading, not compiled, not run on hardware
//
#include <hip/hip_runtime.h>
#include <hip/hip_bf16.h>

typedef __attribute__((ext_vector_type(16))) _Float16 v16h_t;
typedef __attribute__((ext_vector_type(8)))  _Float16 v8h_t;
typedef __attribute__((ext_vector_type(8)))  float    v8f_t;

union V16H { v16h_t v; v8h_t h[2]; _Float16 e[16]; };
union V8H  { v8h_t  v; _Float16 e[8]; };
union V8F  { v8f_t  v; float e[8]; };

#define LDC64 72   // f16 elems per row for 64-ch LDS buffers (144B, 16B aligned)
#define LDC32 40   // f16 elems per row for 32-ch LDS buffers (80B, 16B aligned)

// ---- workspace layout (bytes) ----
constexpr size_t OFF_W1   = 0;        // 9*64*64 f16   = 73728 B
constexpr size_t OFF_WR01 = 73728;    // 9*32*64 f16   = 36864 B
constexpr size_t OFF_WR02 = 110592;   // 64*32 f16     = 4096 B
constexpr size_t OFF_WR11 = 114688;   // 9*32*64 f16   = 36864 B
constexpr size_t OFF_WR12 = 151552;   // 64*32 f16     = 4096 B
constexpr size_t OFF_W2   = 155648;   // 32*64 f16     = 4096 B
constexpr size_t OFF_MASK = 159744;   // 8 samples * 16384 f16 = 262144 B
constexpr size_t OFF_R    = 421888;   // 8 * 32*64 f32 = 65536 B
constexpr size_t OFF_E    = 487424;   // 8 * 64*64 f32 = 131072 B
constexpr size_t OFF_Z    = 618496;   // 8*64 * 32*64 f32 = 4194304 B
// per-sample mask block (f16 elements): mA@0(4096) mB@4096 mC@8192 mH@12288(2048) mG@14336(2048)

__device__ inline v8f_t wmma16(v16h_t a, v16h_t b, v8f_t c) {
  return __builtin_amdgcn_wmma_f32_16x16x32_f16(false, a, false, b, (short)0, c,
                                                false, false);
}

// A fragment: 16x32 f16, row-major [m][k], ldk = K stride. Lane L: m=L%16,
// k = k0 + 16*(L/16) .. +15 (contiguous).
__device__ inline v16h_t load_a_frag(const _Float16* W, int ldk, int m0, int k0, int lane) {
  const int m  = m0 + (lane & 15);
  const int kb = (lane >> 4) << 4;
  const _Float16* p = W + m * ldk + k0 + kb;
  V16H r;
  r.h[0] = *(const v8h_t*)(p);
  r.h[1] = *(const v8h_t*)(p + 8);
  return r.v;
}

// B fragment: 32x16 f16 from activation buffer stored [n][c] (pixel-major,
// channel-contiguous). Lane L: n = n0 + L%16, k = k0 + 16*(L/16) .. +15
// (contiguous). Shift (dy,dx) applied to the 8x8 pixel grid, zero outside.
__device__ inline v16h_t load_b_shift(const _Float16* buf, int ldc, int n0, int k0,
                                      int dy, int dx, int lane) {
  const int n   = n0 + (lane & 15);
  const int pix = n & 63;
  const int tan = n >> 6;
  const int pr = pix >> 3, pc = pix & 7;
  const int sr = pr + dy, sc = pc + dx;
  V16H out;
  if ((unsigned)sr < 8u && (unsigned)sc < 8u) {
    const int q = (tan << 6) + (sr << 3) + sc;
    const _Float16* p = buf + q * ldc + k0 + ((lane >> 4) << 4);
    out.h[0] = *(const v8h_t*)(p);
    out.h[1] = *(const v8h_t*)(p + 8);
  } else {
#pragma unroll
    for (int j = 0; j < 16; ++j) out.e[j] = (_Float16)0.0f;
  }
  return out.v;
}

// B fragment for the analytic dx tangent: nonzero only where shifted pixel == i.
// Tangent 0: all channels = 1.  Tangent 1: channel c = x[b][c][i] (xcol in LDS).
__device__ inline v16h_t load_b_dx(const _Float16* xcol, int ipix, int n0, int k0,
                                   int dy, int dx, int lane) {
  const int n   = n0 + (lane & 15);
  const int pix = n & 63;
  const int tan = n >> 6;
  const int pr = pix >> 3, pc = pix & 7;
  const int sr = pr + dy, sc = pc + dx;
  V16H out;
  const bool hit = ((unsigned)sr < 8u) && ((unsigned)sc < 8u) && (((sr << 3) + sc) == ipix);
  if (hit) {
    if (tan == 0) {
#pragma unroll
      for (int j = 0; j < 16; ++j) out.e[j] = (_Float16)1.0f;
    } else {
      const int kbase = k0 + ((lane >> 4) << 4);
      out.h[0] = *(const v8h_t*)(xcol + kbase);
      out.h[1] = *(const v8h_t*)(xcol + kbase + 8);
    }
  } else {
#pragma unroll
    for (int j = 0; j < 16; ++j) out.e[j] = (_Float16)0.0f;
  }
  return out.v;
}

// One linearized conv stage as WMMA GEMM over N=128 (2 tangents x 64 pixels).
// MODE 0: conv1 : B from dx loader; store t0 -> bufT, maskA*t0 -> bufA
// MODE 1: res3x3: src bufA;         store maskH * u -> bufV
// MODE 2: res1x1: src bufV;         t += w -> bufT, maskB * t -> bufA
// MODE 3: conv2 : src bufA;         store f32 dy -> dyb[tan*32+o][p]
template <int COUT, int CIN, bool K3, int MODE>
__device__ void conv_stage(const _Float16* W,
                           const _Float16* src, int src_ldc,
                           _Float16* bufT, _Float16* bufA, _Float16* bufV,
                           const _Float16* mask, int mask_ldc,
                           float* dyb, const _Float16* xcol, int ipix,
                           int lane, int wave) {
  constexpr int MT = COUT / 16;
  constexpr int NK = CIN / 32;
  constexpr int NS = K3 ? 9 : 1;
  for (int tile = wave; tile < MT * 8; tile += 8) {
    const int mt = tile >> 3;
    const int nt = tile & 7;
    v8f_t acc = {};
    for (int s = 0; s < NS; ++s) {
      const int dy = K3 ? (s / 3 - 1) : 0;
      const int dx = K3 ? (s % 3 - 1) : 0;
      const _Float16* Wb = W + s * (COUT * CIN);
      for (int kb = 0; kb < NK; ++kb) {
        v16h_t a = load_a_frag(Wb, CIN, mt * 16, kb * 32, lane);
        v16h_t bb = (MODE == 0)
                        ? load_b_dx(xcol, ipix, nt * 16, kb * 32, dy, dx, lane)
                        : load_b_shift(src, src_ldc, nt * 16, kb * 32, dy, dx, lane);
        acc = wmma16(a, bb, acc);
      }
    }
    // D tile element (m, n): lane = n%16 + 16*(m/8), vgpr = m%8
    const int n  = nt * 16 + (lane & 15);
    const int p  = n & 63;
    const int m0 = mt * 16 + ((lane >> 4) << 3);
    V8F va; va.v = acc;
    if (MODE == 0 || MODE == 2) {
      V8H mku; mku.v = *(const v8h_t*)(mask + p * mask_ldc + m0);
      V8H th, am;
      if (MODE == 2) {
        V8H told; told.v = *(const v8h_t*)(bufT + n * LDC64 + m0);
#pragma unroll
        for (int j = 0; j < 8; ++j) {
          float tv = va.e[j] + (float)told.e[j];
          th.e[j] = (_Float16)tv;
          am.e[j] = th.e[j] * mku.e[j];
        }
      } else {
#pragma unroll
        for (int j = 0; j < 8; ++j) {
          th.e[j] = (_Float16)va.e[j];
          am.e[j] = th.e[j] * mku.e[j];
        }
      }
      *(v8h_t*)(bufT + n * LDC64 + m0) = th.v;
      *(v8h_t*)(bufA + n * LDC64 + m0) = am.v;
    } else if (MODE == 1) {
      V8H mku; mku.v = *(const v8h_t*)(mask + p * mask_ldc + m0);
      V8H vv;
#pragma unroll
      for (int j = 0; j < 8; ++j) vv.e[j] = ((_Float16)va.e[j]) * mku.e[j];
      *(v8h_t*)(bufV + n * LDC32 + m0) = vv.v;
    } else {  // MODE 3
      const int tan = n >> 6;
#pragma unroll
      for (int j = 0; j < 8; ++j) dyb[(tan * 32 + m0 + j) * 64 + p] = va.e[j];
    }
  }
}

// Online-softmax Hopfield readout for one pixel: toks strided by 64 in LDS/f32.
__device__ inline void hopfield_pixel(const float* toks, const float* patterns,
                                      float* qout) {
  float tok[32];
#pragma unroll
  for (int c = 0; c < 32; ++c) tok[c] = toks[c * 64];
  float q[32];
#pragma unroll
  for (int c = 0; c < 32; ++c) q[c] = 0.f;
  float M = -3.0e38f, S = 0.f;
  for (int e = 0; e < 512; ++e) {
    const float* pe = patterns + e * 32;
    float l = 0.f;
#pragma unroll
    for (int c = 0; c < 32; ++c) l += tok[c] * pe[c];
    l *= 0.17677669529663687f;  // 1/sqrt(32)
    if (l > M) {
      float sc = __expf(M - l);
      S *= sc;
#pragma unroll
      for (int c = 0; c < 32; ++c) q[c] *= sc;
      M = l;
    }
    float w = __expf(l - M);
    S += w;
#pragma unroll
    for (int c = 0; c < 32; ++c) q[c] += w * pe[c];
  }
  float invS = 1.f / S;
#pragma unroll
  for (int c = 0; c < 32; ++c) qout[c] = q[c] * invS;
}

// ---------------- kernel 0: repack weights to f16 [shift][co][ci] ----------------
__global__ __launch_bounds__(256)
void prep_weights(const float* w1, const float* r01, const float* r02,
                  const float* r11, const float* r12, const float* w2,
                  _Float16* W1, _Float16* WR01, _Float16* WR02,
                  _Float16* WR11, _Float16* WR12, _Float16* W2h) {
  const int t = blockIdx.x * blockDim.x + threadIdx.x;
  if (t < 36864) {  // conv1 9*64*64
    int s = t / 4096, rem = t % 4096, co = rem >> 6, ci = rem & 63;
    W1[t] = (_Float16)w1[((co * 64 + ci) * 3 + (s / 3)) * 3 + (s % 3)];
  }
  if (t < 18432) {  // res 3x3: 9*32*64
    int s = t / 2048, rem = t % 2048, co = rem >> 6, ci = rem & 63;
    int src = ((co * 64 + ci) * 3 + (s / 3)) * 3 + (s % 3);
    WR01[t] = (_Float16)r01[src];
    WR11[t] = (_Float16)r11[src];
  }
  if (t < 2048) {   // 1x1 weights, already [co][ci]
    WR02[t] = (_Float16)r02[t];
    WR12[t] = (_Float16)r12[t];
    W2h[t]  = (_Float16)w2[t];
  }
}

// ---------------- kernel 1: primal forward, masks, residual r -------------------
__global__ __launch_bounds__(256)
void primal_kernel(const float* x, const float* w1, const float* b1,
                   const float* r01w, const float* r02w,
                   const float* r11w, const float* r12w,
                   const float* w2, const float* b2, const float* patterns,
                   _Float16* maskbase, float* rbase) {
  const int b = blockIdx.x, tid = threadIdx.x;
  __shared__ float bufX[64 * 64];  // relu'd activations [c][p]
  __shared__ float bufY[64 * 64];  // running y  [c][p]
  __shared__ float bufH[32 * 64];
  __shared__ float bufO[32 * 64];
  _Float16* mk = maskbase + b * 16384;
  float* rws = rbase + b * 2048;

  for (int t = tid; t < 4096; t += 256) bufX[t] = x[b * 4096 + t];
  __syncthreads();
  // conv1 3x3 + bias -> bufY
  for (int t = tid; t < 4096; t += 256) {
    int co = t >> 6, p = t & 63, pr = p >> 3, pc = p & 7;
    float acc = b1[co];
    for (int ci = 0; ci < 64; ++ci) {
      const float* wr = w1 + (co * 64 + ci) * 9;
      const float* xr = bufX + ci * 64;
      for (int ky = 0; ky < 3; ++ky) {
        int sr = pr + ky - 1; if ((unsigned)sr >= 8u) continue;
        for (int kx = 0; kx < 3; ++kx) {
          int sc = pc + kx - 1; if ((unsigned)sc >= 8u) continue;
          acc += wr[ky * 3 + kx] * xr[sr * 8 + sc];
        }
      }
    }
    bufY[t] = acc;
  }
  __syncthreads();
  for (int t = tid; t < 4096; t += 256) {  // mask A, relu
    float v = bufY[t]; int c = t >> 6, p = t & 63;
    mk[p * 64 + c] = (v > 0.f) ? (_Float16)1.0f : (_Float16)0.0f;
    bufX[t] = v > 0.f ? v : 0.f;
  }
  __syncthreads();
  for (int blk = 0; blk < 2; ++blk) {  // two residual blocks
    const float* wa = blk ? r11w : r01w;
    const float* wb = blk ? r12w : r02w;
    const int moffH = blk ? 14336 : 12288;   // mH / mG
    const int moffB = blk ? 8192  : 4096;    // mC / mB
    for (int t = tid; t < 2048; t += 256) {  // 3x3, 64->32
      int co = t >> 6, p = t & 63, pr = p >> 3, pc = p & 7;
      float acc = 0.f;
      for (int ci = 0; ci < 64; ++ci) {
        const float* wr = wa + (co * 64 + ci) * 9;
        const float* xr = bufX + ci * 64;
        for (int ky = 0; ky < 3; ++ky) {
          int sr = pr + ky - 1; if ((unsigned)sr >= 8u) continue;
          for (int kx = 0; kx < 3; ++kx) {
            int sc = pc + kx - 1; if ((unsigned)sc >= 8u) continue;
            acc += wr[ky * 3 + kx] * xr[sr * 8 + sc];
          }
        }
      }
      bufH[t] = acc;
    }
    __syncthreads();
    for (int t = tid; t < 2048; t += 256) {
      float v = bufH[t]; int c = t >> 6, p = t & 63;
      mk[moffH + p * 32 + c] = (v > 0.f) ? (_Float16)1.0f : (_Float16)0.0f;
      bufH[t] = fmaxf(v, 0.f);
    }
    __syncthreads();
    for (int t = tid; t < 4096; t += 256) {  // 1x1 32->64 + residual
      int co = t >> 6, p = t & 63;
      float acc = 0.f;
      for (int ci = 0; ci < 32; ++ci) acc += wb[co * 32 + ci] * bufH[ci * 64 + p];
      float yn = bufY[t] + acc;
      bufY[t] = yn;
      mk[moffB + p * 64 + co] = (yn > 0.f) ? (_Float16)1.0f : (_Float16)0.0f;
      bufX[t] = fmaxf(yn, 0.f);
    }
    __syncthreads();
  }
  for (int t = tid; t < 2048; t += 256) {  // conv2 1x1 + bias
    int co = t >> 6, p = t & 63;
    float acc = b2[co];
    for (int ci = 0; ci < 64; ++ci) acc += w2[co * 64 + ci] * bufX[ci * 64 + p];
    bufO[t] = acc;
  }
  __syncthreads();
  if (tid < 64) {  // hopfield + residual r = 2*(y - yq)
    float q[32];
    hopfield_pixel(bufO + tid, patterns, q);
#pragma unroll
    for (int o = 0; o < 32; ++o)
      rws[o * 64 + tid] = 2.f * (bufO[o * 64 + tid] - q[o]);
  }
}

// ---------------- kernel 2: WMMA tangent passes (the heavy one) -----------------
__global__ __launch_bounds__(256)
void tangent_kernel(const float* x, char* ws) {
  const int i = blockIdx.x;   // input pixel
  const int b = blockIdx.y;   // sample
  const int tid = threadIdx.x;
  const int lane = tid & 31;
  const int wave = tid >> 5;

  __shared__ __align__(16) _Float16 bufT[128 * LDC64];  // running tangent t (64ch)
  __shared__ __align__(16) _Float16 bufA[128 * LDC64];  // masked conv input (64ch)
  __shared__ __align__(16) _Float16 bufV[128 * LDC32];  // 32ch intermediate
  __shared__ __align__(16) float dyb[2 * 32 * 64];      // final dy [tan][o][p]
  __shared__ __align__(16) _Float16 xcol[64];

  const _Float16* W1   = (const _Float16*)(ws + OFF_W1);
  const _Float16* WR01 = (const _Float16*)(ws + OFF_WR01);
  const _Float16* WR02 = (const _Float16*)(ws + OFF_WR02);
  const _Float16* WR11 = (const _Float16*)(ws + OFF_WR11);
  const _Float16* WR12 = (const _Float16*)(ws + OFF_WR12);
  const _Float16* W2h  = (const _Float16*)(ws + OFF_W2);
  const _Float16* mk   = (const _Float16*)(ws + OFF_MASK) + b * 16384;
  const float* rws     = (const float*)(ws + OFF_R) + b * 2048;
  float* e_total       = (float*)(ws + OFF_E);
  float* zbuf          = (float*)(ws + OFF_Z);

  if (tid < 64) xcol[tid] = (_Float16)x[(b * 64 + tid) * 64 + i];
  __syncthreads();

  conv_stage<64, 64, true,  0>(W1,   nullptr, 0,     bufT, bufA, bufV, mk,          64, dyb, xcol, i, lane, wave);
  __syncthreads();
  conv_stage<32, 64, true,  1>(WR01, bufA, LDC64, bufT, bufA, bufV, mk + 12288, 32, dyb, xcol, i, lane, wave);
  __syncthreads();
  conv_stage<64, 32, false, 2>(WR02, bufV, LDC32, bufT, bufA, bufV, mk + 4096,  64, dyb, xcol, i, lane, wave);
  __syncthreads();
  conv_stage<32, 64, true,  1>(WR11, bufA, LDC64, bufT, bufA, bufV, mk + 14336, 32, dyb, xcol, i, lane, wave);
  __syncthreads();
  conv_stage<64, 32, false, 2>(WR12, bufV, LDC32, bufT, bufA, bufV, mk + 8192,  64, dyb, xcol, i, lane, wave);
  __syncthreads();
  conv_stage<32, 64, false, 3>(W2h,  bufA, LDC64, bufT, bufA, bufV, nullptr,     0, dyb, xcol, i, lane, wave);
  __syncthreads();

  if (tid < 64) {  // e_total[b][m][i] from ones-tangent (dyb tan 0)
    float e = 0.f;
#pragma unroll
    for (int o = 0; o < 32; ++o) e += rws[o * 64 + tid] * dyb[o * 64 + tid];
    e_total[(b * 64 + tid) * 64 + i] = e;
  }
  float* zb = zbuf + (size_t)(b * 64 + i) * 2048;  // z_i[o][m] from x-tangent
  for (int t = tid; t < 2048; t += 256) zb[t] = dyb[2048 + t];
}

// ---------------- kernel 3: argmin routing, scatter, final hopfield --------------
__global__ __launch_bounds__(256)
void finalize_kernel(const float* e_total, const float* z, const float* patterns,
                     float* out) {
  const int b = blockIdx.x, tid = threadIdx.x;
  __shared__ int idx[64];
  __shared__ float ym[32 * 64];
  if (tid < 64) {
    const float* e = e_total + b * 4096;  // [m][i]
    float best = e[tid];
    int bi = 0;
    for (int m = 1; m < 64; ++m) {
      float v = e[m * 64 + tid];
      if (v < best) { best = v; bi = m; }
    }
    idx[tid] = bi;
  }
  for (int t = tid; t < 2048; t += 256) ym[t] = 0.f;
  __syncthreads();
  if (tid < 32) {
    const int o = tid;
    for (int i = 0; i < 64; ++i) {
      int m = idx[i];
      ym[o * 64 + m] += z[(size_t)(b * 64 + i) * 2048 + o * 64 + m];
    }
  }
  __syncthreads();
  if (tid < 64) {
    float q[32];
    hopfield_pixel(ym + tid, patterns, q);
#pragma unroll
    for (int o = 0; o < 32; ++o) out[(b * 32 + o) * 64 + tid] = q[o];
  }
}

extern "C" void kernel_launch(void* const* d_in, const int* in_sizes, int n_in,
                              void* d_out, int out_size, void* d_ws, size_t ws_size,
                              hipStream_t stream) {
  (void)in_sizes; (void)n_in; (void)out_size; (void)ws_size;
  const float* x        = (const float*)d_in[0];
  const float* conv1_w  = (const float*)d_in[1];
  const float* conv1_b  = (const float*)d_in[2];
  const float* res0_w1  = (const float*)d_in[3];
  const float* res0_w2  = (const float*)d_in[4];
  const float* res1_w1  = (const float*)d_in[5];
  const float* res1_w2  = (const float*)d_in[6];
  const float* conv2_w  = (const float*)d_in[7];
  const float* conv2_b  = (const float*)d_in[8];
  const float* patterns = (const float*)d_in[9];
  float* out = (float*)d_out;
  char* ws = (char*)d_ws;

  prep_weights<<<144, 256, 0, stream>>>(
      conv1_w, res0_w1, res0_w2, res1_w1, res1_w2, conv2_w,
      (_Float16*)(ws + OFF_W1), (_Float16*)(ws + OFF_WR01),
      (_Float16*)(ws + OFF_WR02), (_Float16*)(ws + OFF_WR11),
      (_Float16*)(ws + OFF_WR12), (_Float16*)(ws + OFF_W2));

  primal_kernel<<<8, 256, 0, stream>>>(
      x, conv1_w, conv1_b, res0_w1, res0_w2, res1_w1, res1_w2,
      conv2_w, conv2_b, patterns,
      (_Float16*)(ws + OFF_MASK), (float*)(ws + OFF_R));

  tangent_kernel<<<dim3(64, 8), 256, 0, stream>>>(x, ws);

  finalize_kernel<<<8, 256, 0, stream>>>(
      (const float*)(ws + OFF_E), (const float*)(ws + OFF_Z), patterns, out);
}
